// SS2D_44143673868981
// MI455X (gfx1250) — compile-verified, matrix operand order checked by reading, not run on hardware
//
#include <hip/hip_runtime.h>
#include <hip/hip_bf16.h>

// ---------------- problem constants ----------------
#define BB     2
#define HH     48
#define WW     48
#define LL     (HH * WW)            // 2304
#define DMODEL 192
#define CIN    (2 * DMODEL)         // 384 input channels to in_proj
#define CC     768                  // C_INNER
#define KG     4
#define DD     (CC / KG)            // 192
#define NS     16
#define RR     12
#define CDBL   (RR + 2 * NS)        // 44
#define NPOS   (BB * LL)            // 4608

typedef __attribute__((ext_vector_type(16))) __bf16 v16bf;
typedef __attribute__((ext_vector_type(8)))  float  v8f;
typedef __attribute__((ext_vector_type(4)))  unsigned int v4u;
typedef __attribute__((ext_vector_type(8)))  int v8i;
typedef __attribute__((ext_vector_type(4)))  int v4i;

#if defined(__has_builtin)
#if __has_builtin(__builtin_amdgcn_tensor_load_to_lds) && \
    __has_builtin(__builtin_amdgcn_s_wait_tensorcnt)
#define USE_TDM 1
#endif
#endif

union FragAB {
    v16bf v;
    unsigned short us[16];
    unsigned int   ui[8];
};

__device__ __forceinline__ unsigned short f32_to_bf16(float f) {
    unsigned int u = __float_as_uint(f);
    u = (u + 0x7FFFu + ((u >> 16) & 1u)) >> 16;
    return (unsigned short)u;
}

// K start of packed bf16 pair held by slot j (0..7) for a lane in half `half`
// (ISA 7.12.2, 16-bit A-matrix 16x32 layout; mirrored for B with N across lanes)
__device__ __forceinline__ int frag_k0(int j, int half) {
    int base = (j < 4) ? 0 : 16;
    return base + half * 8 + 2 * (j & 3);
}

// ---------------- cast f32 -> bf16 ----------------
__global__ void cast_bf16_kernel(const float* __restrict__ in,
                                 unsigned short* __restrict__ out, int n) {
    int i = blockIdx.x * blockDim.x + threadIdx.x;
    if (i < n) out[i] = f32_to_bf16(in[i]);
}

// ---------------- WMMA GEMM: C[M,N] = A[M,Kd] * B^T  (Bw stored [N,Kd]) ------
// one wave per 16x16 output tile
__global__ void gemm_nt_bf16(const unsigned short* __restrict__ A,
                             const unsigned short* __restrict__ Bw,
                             float* __restrict__ C,
                             int M, int N, int Kd) {
    int wave  = (blockIdx.x * blockDim.x + threadIdx.x) >> 5;
    int lane  = threadIdx.x & 31;
    int tilesN = N >> 4;
    int tm = wave / tilesN;
    int tn = wave - tm * tilesN;
    if (tm * 16 >= M) return;
    int m0 = tm << 4, n0 = tn << 4;
    int col  = lane & 15;
    int half = lane >> 4;

    const unsigned short* Ar = A  + (m0 + col) * Kd;   // lane's A row
    const unsigned short* Br = Bw + (n0 + col) * Kd;   // lane's B column (= W row)
    v8f acc = {};

    for (int kb = 0; kb < Kd; kb += 32) {
        __builtin_prefetch(Ar + kb + 64, 0, 0);
        __builtin_prefetch(Br + kb + 64, 0, 0);
        FragAB a, b;
#pragma unroll
        for (int j = 0; j < 8; ++j) {
            int k0 = frag_k0(j, half);
            a.ui[j] = *(const unsigned int*)(Ar + kb + k0);
            b.ui[j] = *(const unsigned int*)(Br + kb + k0);
        }
        acc = __builtin_amdgcn_wmma_f32_16x16x32_bf16(
            false, a.v, false, b.v, (short)0, acc, false, false);
    }
#pragma unroll
    for (int r = 0; r < 8; ++r)
        C[(m0 + r + 8 * half) * N + n0 + col] = acc[r];
}

// ---------------- depthwise conv3x3 + SiLU + 4-direction scatter -------------
__global__ void conv_silu_reorder(const float* __restrict__ xz,   // [B,H,W,CC]
                                  const float* __restrict__ cw,   // [CC,1,3,3]
                                  const float* __restrict__ cb,   // [CC]
                                  float* __restrict__ xs,         // [B,K,DD,L]
                                  unsigned short* __restrict__ xsb) {
    int idx = blockIdx.x * blockDim.x + threadIdx.x;
    if (idx >= BB * CC * HH * WW) return;
    int w = idx % WW;  int t = idx / WW;
    int h = t % HH;    t /= HH;
    int c = t % CC;    int b = t / CC;

    float acc = cb[c];
#pragma unroll
    for (int i = 0; i < 3; ++i) {
        int hh = h + i - 1;
        if (hh < 0 || hh >= HH) continue;
#pragma unroll
        for (int j = 0; j < 3; ++j) {
            int ww = w + j - 1;
            if (ww < 0 || ww >= WW) continue;
            acc += xz[((b * HH + hh) * WW + ww) * CC + c] * cw[c * 9 + i * 3 + j];
        }
    }
    float v = acc / (1.f + __expf(-acc));   // SiLU

    int k = c / DD, dd = c % DD;
    int l0 = h * WW + w;
    int l1 = w * HH + h;
    int lk = (k == 0) ? l0 : (k == 1) ? l1 : (k == 2) ? (LL - 1 - l0) : (LL - 1 - l1);
    int o  = ((b * KG + k) * DD + dd) * LL + lk;
    xs[o]  = v;
    xsb[o] = f32_to_bf16(v);
}

// ---------------- WMMA GEMM x_proj: Xd[bk,44,L] = Wp[k,44,192] * Xs[bk,192,L] -
// TDM path: each wave DMA-stages its 192x16 B tile (row stride L) into LDS via
// tensor_load_to_lds, waits on TENSORcnt, then feeds WMMA from LDS.
__global__ void gemm_xproj(const unsigned short* __restrict__ Wp,  // [KG,44,192] bf16
                           const unsigned short* __restrict__ Xs,  // [B*KG,192,L] bf16
                           float* __restrict__ Xd) {               // [B*KG,44,L]
    const int tilesM = 3, tilesN = LL >> 4;                         // 3 x 144
#ifdef USE_TDM
    __shared__ unsigned short smem[4][DD * 16];                     // 4 waves * 6 KB
#endif
    int wave = __builtin_amdgcn_readfirstlane(
        (int)((blockIdx.x * blockDim.x + threadIdx.x) >> 5));
    int wib  = __builtin_amdgcn_readfirstlane((int)(threadIdx.x >> 5)) & 3;
    int lane = threadIdx.x & 31;
    int bk   = wave / (tilesM * tilesN);
    if (bk >= BB * KG) return;
    int rem  = wave - bk * (tilesM * tilesN);
    int tm = rem / tilesN, tn = rem - tm * tilesN;
    int m0 = tm << 4, n0 = tn << 4;
    int col = lane & 15, half = lane >> 4;
    int kgr = bk & 3;

    v8f acc = {};
    int am = m0 + col;   // A row for this lane (guard m < 44)
    const unsigned short* Ar = Wp + (kgr * CDBL + (am < CDBL ? am : 0)) * DD;

#ifdef USE_TDM
    // ---- Tensor DMA: stage B tile Xs[bk, 0:192, n0:n0+16] into LDS ----
    {
        unsigned long long ga =
            (unsigned long long)(const void*)(Xs + (size_t)bk * DD * LL + n0);
        unsigned int lds_off = (unsigned int)(wib * (DD * 16 * 2));  // byte offset
        v4u g0;
        g0[0] = 1u;                                   // count=1, user mode, no gather
        g0[1] = lds_off;                              // lds_addr
        g0[2] = (unsigned int)(ga & 0xFFFFFFFFull);   // global_addr[31:0]
        g0[3] = (unsigned int)((ga >> 32) & 0x1FFFFFFull) | (2u << 30); // addr hi | type=2
        v8i g1;
        g1[0] = (int)(1u << 16);        // workgroup_mask=0, data_size=1 (2 bytes)
        g1[1] = (int)((unsigned)LL << 16);            // tensor_dim0[15:0] at bits 63:48
        g1[2] = (int)((unsigned)DD << 16);            // dim0 hi=0 | tensor_dim1[15:0]
        g1[3] = (int)(16u << 16);                     // dim1 hi=0 | tile_dim0=16
        g1[4] = (int)DD;                              // tile_dim1=192 | tile_dim2=0
        g1[5] = (int)LL;                              // tensor_dim0_stride lo32
        g1[6] = 0;                                    // stride hi | dim1_stride lo
        g1[7] = 0;
        v4i g2 = {0, 0, 0, 0};
        v4i g3 = {0, 0, 0, 0};
        v8i g4 = {0, 0, 0, 0, 0, 0, 0, 0};            // extra group (6-arg toolchain)
        __builtin_amdgcn_tensor_load_to_lds(g0, g1, g2, g3, g4, 0);
        __builtin_amdgcn_s_wait_tensorcnt(0);
    }
#else
    const unsigned short* Bb = Xs + (bk * DD) * LL + n0 + col;
#endif

    for (int kb = 0; kb < DD; kb += 32) {
        FragAB a, b;
#pragma unroll
        for (int j = 0; j < 8; ++j) {
            int k0 = frag_k0(j, half);
            unsigned int av = *(const unsigned int*)(Ar + kb + k0);
            a.ui[j] = (am < CDBL) ? av : 0u;
#ifdef USE_TDM
            unsigned int lo = smem[wib][(kb + k0) * 16 + col];
            unsigned int hi = smem[wib][(kb + k0 + 1) * 16 + col];
#else
            unsigned int lo = Bb[(kb + k0) * LL];
            unsigned int hi = Bb[(kb + k0 + 1) * LL];
#endif
            b.ui[j] = lo | (hi << 16);
        }
        acc = __builtin_amdgcn_wmma_f32_16x16x32_bf16(
            false, a.v, false, b.v, (short)0, acc, false, false);
    }
#pragma unroll
    for (int r = 0; r < 8; ++r) {
        int m = m0 + r + 8 * half;
        if (m < CDBL)
            Xd[(bk * CDBL + m) * LL + n0 + col] = acc[r];
    }
}

// ---------------- dt projection (K=12, VALU) + softplus ----------------------
__global__ void dt_softplus(const float* __restrict__ xdbl,  // [BK,44,L]
                            const float* __restrict__ dtw,   // [KG,DD,RR]
                            const float* __restrict__ dtb,   // [KG,DD]
                            float* __restrict__ dts) {       // [BK,DD,L]
    int idx = blockIdx.x * blockDim.x + threadIdx.x;
    if (idx >= BB * KG * DD * LL) return;
    int l = idx % LL;  int t = idx / LL;
    int dd = t % DD;   int bk = t / DD;  int k = bk & 3;

    float acc = dtb[k * DD + dd];
    const float* wr = dtw + (k * DD + dd) * RR;
    const float* xb = xdbl + (bk * CDBL) * LL + l;
#pragma unroll
    for (int r = 0; r < RR; ++r)
        acc += xb[r * LL] * wr[r];
    dts[idx] = (acc > 20.f) ? acc : log1pf(__expf(acc));
}

// ---------------- selective scan: 1 thread per (b,k,d), 16 states in regs ----
__global__ void selective_scan(const float* __restrict__ xs,    // [BK,DD,L] (u)
                               const float* __restrict__ dts,   // [BK,DD,L]
                               const float* __restrict__ xdbl,  // [BK,44,L]
                               const float* __restrict__ A_logs,// [KG*DD,NS]
                               const float* __restrict__ Ds,    // [KG*DD]
                               float* __restrict__ ys) {        // [BK,DD,L]
    int t = blockIdx.x * blockDim.x + threadIdx.x;
    if (t >= BB * KG * DD) return;
    int dd = t % DD, bk = t / DD, k = bk & 3;

    float Av[NS], h[NS];
#pragma unroll
    for (int n = 0; n < NS; ++n) {
        Av[n] = -__expf(A_logs[(k * DD + dd) * NS + n]);
        h[n] = 0.f;
    }
    float Dv = Ds[k * DD + dd];

    const float* up = xs  + (bk * DD + dd) * LL;
    const float* dp = dts + (bk * DD + dd) * LL;
    const float* Bp = xdbl + (bk * CDBL + RR) * LL;
    const float* Cp = xdbl + (bk * CDBL + RR + NS) * LL;
    float* yp = ys + (bk * DD + dd) * LL;

    for (int l = 0; l < LL; ++l) {
        float u  = up[l];
        float dt = dp[l];
        float du = dt * u;
        float y = 0.f;
#pragma unroll
        for (int n = 0; n < NS; ++n) {
            float Bn = Bp[n * LL + l];
            float Cn = Cp[n * LL + l];
            h[n] = h[n] * __expf(dt * Av[n]) + du * Bn;
            y += h[n] * Cn;
        }
        yp[l] = y + Dv * u;
    }
}

// ---------------- inverse reorder + LayerNorm(768) + bf16 cast --------------
__global__ void merge_norm(const float* __restrict__ ys,  // [BK,DD,L]
                           const float* __restrict__ nw,  // [CC]
                           const float* __restrict__ nb,  // [CC]
                           unsigned short* __restrict__ yb) { // [NPOS,CC] bf16
    __shared__ float ssum[256];
    __shared__ float ssq[256];
    int pos = blockIdx.x;              // 0..NPOS-1
    int b = pos / LL;
    int l = pos % LL;
    int h = l / WW, w = l % WW;
    int l0 = h * WW + w;
    int l1 = w * HH + h;

    float vals[3];
    float s = 0.f, s2 = 0.f;
#pragma unroll
    for (int i = 0; i < 3; ++i) {
        int c = threadIdx.x + i * 256;
        int k = c / DD, dd = c % DD;
        int lk = (k == 0) ? l0 : (k == 1) ? l1 : (k == 2) ? (LL - 1 - l0) : (LL - 1 - l1);
        float v = ys[((b * KG + k) * DD + dd) * LL + lk];
        vals[i] = v;
        s += v; s2 += v * v;
    }
    ssum[threadIdx.x] = s;
    ssq[threadIdx.x]  = s2;
    __syncthreads();
    for (int st = 128; st > 0; st >>= 1) {
        if (threadIdx.x < st) {
            ssum[threadIdx.x] += ssum[threadIdx.x + st];
            ssq[threadIdx.x]  += ssq[threadIdx.x + st];
        }
        __syncthreads();
    }
    float mu  = ssum[0] * (1.f / CC);
    float var = ssq[0] * (1.f / CC) - mu * mu;
    float rs  = rsqrtf(var + 1e-5f);
#pragma unroll
    for (int i = 0; i < 3; ++i) {
        int c = threadIdx.x + i * 256;
        float o = (vals[i] - mu) * rs * nw[c] + nb[c];
        yb[pos * CC + c] = f32_to_bf16(o);
    }
}

// ---------------- host launch ----------------
extern "C" void kernel_launch(void* const* d_in, const int* in_sizes, int n_in,
                              void* d_out, int out_size, void* d_ws, size_t ws_size,
                              hipStream_t stream) {
    const float* x        = (const float*)d_in[0];   // [B,H,W,2*DMODEL]
    const float* in_w     = (const float*)d_in[1];   // [CC, CIN]
    const float* conv_w   = (const float*)d_in[2];   // [CC,1,3,3]
    const float* conv_b   = (const float*)d_in[3];   // [CC]
    const float* xproj_w  = (const float*)d_in[4];   // [KG,44,DD]
    const float* dt_w     = (const float*)d_in[5];   // [KG,DD,RR]
    const float* dt_b     = (const float*)d_in[6];   // [KG,DD]
    const float* A_logs   = (const float*)d_in[7];   // [KG*DD,NS]
    const float* Ds       = (const float*)d_in[8];   // [KG*DD]
    const float* norm_w   = (const float*)d_in[9];   // [CC]
    const float* norm_b   = (const float*)d_in[10];  // [CC]
    const float* out_w    = (const float*)d_in[11];  // [2*DMODEL, CC]
    float* out = (float*)d_out;                      // [NPOS, 2*DMODEL] f32

    // workspace layout
    float* ws   = (float*)d_ws;
    float* xz   = ws;                       // NPOS*CC
    float* xs   = xz   + (size_t)NPOS * CC; // BK*DD*LL  (== NPOS*CC)
    float* xdbl = xs   + (size_t)NPOS * CC; // BK*44*LL
    float* dts  = xdbl + (size_t)BB * KG * CDBL * LL;
    float* ysb  = dts  + (size_t)NPOS * CC;
    unsigned short* xb  = (unsigned short*)(ysb + (size_t)NPOS * CC); // x bf16 NPOS*CIN
    unsigned short* wib = xb  + (size_t)NPOS * CIN;                   // in_w bf16 CC*CIN
    unsigned short* xpb = wib + (size_t)CC * CIN;                     // xproj bf16 KG*44*DD
    unsigned short* wob = xpb + (size_t)KG * CDBL * DD;               // out_w bf16 (2*DMODEL)*CC
    unsigned short* xsb = wob + (size_t)2 * DMODEL * CC;              // xs bf16
    unsigned short* yb  = xsb + (size_t)NPOS * CC;                    // y bf16 NPOS*CC

    // 1) casts to bf16
    {
        int n;
        n = NPOS * CIN;
        cast_bf16_kernel<<<(n + 255) / 256, 256, 0, stream>>>(x, xb, n);
        n = CC * CIN;
        cast_bf16_kernel<<<(n + 255) / 256, 256, 0, stream>>>(in_w, wib, n);
        n = KG * CDBL * DD;
        cast_bf16_kernel<<<(n + 255) / 256, 256, 0, stream>>>(xproj_w, xpb, n);
        n = 2 * DMODEL * CC;
        cast_bf16_kernel<<<(n + 255) / 256, 256, 0, stream>>>(out_w, wob, n);
    }

    // 2) in_proj: xz[NPOS,CC] = x[NPOS,CIN] * in_w^T
    {
        int waves = (NPOS / 16) * (CC / 16);           // 288*48
        gemm_nt_bf16<<<waves / 4, 128, 0, stream>>>(xb, wib, xz, NPOS, CC, CIN);
    }

    // 3) depthwise conv + SiLU + 4-direction scatter
    {
        int n = BB * CC * HH * WW;
        conv_silu_reorder<<<(n + 255) / 256, 256, 0, stream>>>(xz, conv_w, conv_b, xs, xsb);
    }

    // 4) x_proj: xdbl[BK,44,L] = xproj_w[k] * xs[bk]   (TDM-staged B tiles)
    {
        int waves = BB * KG * 3 * (LL / 16);           // 8*3*144 = 3456
        gemm_xproj<<<waves / 4, 128, 0, stream>>>(xpb, xsb, xdbl);
    }

    // 5) dt projection + softplus
    {
        int n = BB * KG * DD * LL;
        dt_softplus<<<(n + 255) / 256, 256, 0, stream>>>(xdbl, dt_w, dt_b, dts);
    }

    // 6) selective scan
    {
        int n = BB * KG * DD;                          // 1536 threads
        selective_scan<<<(n + 255) / 256, 256, 0, stream>>>(xs, dts, xdbl, A_logs, Ds, ysb);
    }

    // 7) merge + LayerNorm + bf16
    merge_norm<<<NPOS, 256, 0, stream>>>(ysb, norm_w, norm_b, yb);

    // 8) out_proj: out[NPOS, 2*DMODEL] = y[NPOS,CC] * out_w^T
    {
        int waves = (NPOS / 16) * ((2 * DMODEL) / 16); // 288*24
        gemm_nt_bf16<<<waves / 4, 128, 0, stream>>>(yb, wob, out, NPOS, 2 * DMODEL, CC);
    }
    (void)in_sizes; (void)n_in; (void)out_size; (void)ws_size;
}
